// GNN_73263552135826
// MI455X (gfx1250) — matrix operand order repaired and checked by reading, not compiled, not run on hardware
//
#include <hip/hip_runtime.h>
#include <stdint.h>

// ---------------------------------------------------------------------------
// CDNA5 (gfx1250) GraphNet pipeline.
//
// Heavy GEMMs (edge-MLP / node1-MLP over 400K rows, H=128) are mapped to
// v_wmma_f32_16x16x32_f16: A tiles assembled (with fused gather) into LDS as
// f16, B tiles pre-packed into the ISA fragment layout, f32 accumulation.
// scatter_mean is fused into the node1 epilogue via global_atomic_add_f32.
// ---------------------------------------------------------------------------

typedef __attribute__((ext_vector_type(16))) _Float16 v16h;
typedef __attribute__((ext_vector_type(8)))  float    v8f;

#define NTHREADS 256

struct Seg {
    const float* src;
    const int*   i1;   // optional first-level index (row/col/batch), may be null
    const int*   i2;   // optional second-level index (batch[...]), may be null
    int          w;    // feature width of this segment
};

struct MlpArgs {
    Seg  seg[4];
    int  nseg;
    int  din, dinp;          // din and din padded to multiple of 32
    int  dout;               // 128 or 1
    const _Float16* w1p;     // packed W1 fragments
    const float*    b1;
    const _Float16* w2p;     // packed W2 fragments (dout==128)
    const float*    w2raw;   // raw fp32 W2 (dout==1)
    const float*    b2;
    float*          out;
    const int*      scat;    // if non-null: atomicAdd into out[scat[row]*128+n]
    int             R;       // number of valid rows
};

// ---------------------------------------------------------------------------
// Generic fused gather -> MLP(relu) -> store/scatter kernel.
// Block = 256 threads (8 waves), one block per 16-row tile.
// ---------------------------------------------------------------------------
__global__ __launch_bounds__(NTHREADS) void mlp_kernel(MlpArgs a) {
    __shared__ _Float16 ldsA[16 * 512];   // input tile (f16), din <= 512
    __shared__ _Float16 ldsH[16 * 128];   // hidden tile (f16)

    const int tid  = threadIdx.x;
    const int lane = tid & 31;
    const int wid  = tid >> 5;            // 0..7 -> N-chunk of 16 columns
    const int row0 = blockIdx.x * 16;

    const int dinp = a.dinp;

    // ---- assemble A tile (16 x dinp) into LDS with fused gather ----
    for (int idx = tid; idx < 16 * dinp; idx += NTHREADS) {
        const int m = idx / dinp;
        const int k = idx - m * dinp;
        float v = 0.0f;
        const int rg = row0 + m;
        if (rg < a.R && k < a.din) {
            int off = 0;
            for (int s = 0; s < a.nseg; ++s) {
                const int w = a.seg[s].w;
                if (k < off + w) {
                    int r1 = a.seg[s].i1 ? a.seg[s].i1[rg] : rg;
                    int r2 = a.seg[s].i2 ? a.seg[s].i2[r1] : r1;
                    v = a.seg[s].src[(long)r2 * w + (k - off)];
                    break;
                }
                off += w;
            }
        }
        ldsA[m * dinp + k] = (_Float16)v;
    }
    __syncthreads();

    // A-fragment K interleave (ISA 7.12.2, 16-bit A 16x32):
    // lanes 0-15 hold K {0..7,16..23}, lanes 16-31 hold K {8..15,24..31}.
    const int khb  = (lane >> 4) << 3;    // +8 for upper half-wave (also C M-offset)
    const int mrow = lane & 15;

    // ---- GEMM1: hidden(16x128) = A(16,dinp) x W1(dinp,128) ----
    v8f c = {};
    const int kt = dinp >> 5;
    const v16h* w1 = (const v16h*)a.w1p;
    for (int t = 0; t < kt; ++t) {
        v16h av;
#pragma unroll
        for (int vv = 0; vv < 8; ++vv) {
            const int kb = ((vv < 4) ? (2 * vv) : (16 + 2 * (vv - 4))) + khb;
            const _Float16* p = &ldsA[mrow * dinp + t * 32 + kb];
            av[2 * vv]     = p[0];
            av[2 * vv + 1] = p[1];
        }
        const v16h bv = w1[(wid * kt + t) * 32 + lane];
        c = __builtin_amdgcn_wmma_f32_16x16x32_f16(false, av, false, bv,
                                                   (short)0, c, false, false);
    }

    // ---- bias + relu -> ldsH ----
    {
        const int n  = wid * 16 + (lane & 15);
        const float bb = a.b1[n];
#pragma unroll
        for (int r = 0; r < 8; ++r) {
            const int m = r + khb;                 // C layout: M = r + 8*(lane>=16)
            float v = c[r] + bb;
            ldsH[m * 128 + n] = (_Float16)(v > 0.0f ? v : 0.0f);
        }
    }
    __syncthreads();

    if (a.dout == 128) {
        // ---- GEMM2: out(16x128) = H(16,128) x W2(128,128) ----
        v8f c2 = {};
        const v16h* w2 = (const v16h*)a.w2p;
        for (int t = 0; t < 4; ++t) {
            v16h av;
#pragma unroll
            for (int vv = 0; vv < 8; ++vv) {
                const int kb = ((vv < 4) ? (2 * vv) : (16 + 2 * (vv - 4))) + khb;
                const _Float16* p = &ldsH[mrow * 128 + t * 32 + kb];
                av[2 * vv]     = p[0];
                av[2 * vv + 1] = p[1];
            }
            const v16h bv = w2[(wid * 4 + t) * 32 + lane];
            c2 = __builtin_amdgcn_wmma_f32_16x16x32_f16(false, av, false, bv,
                                                        (short)0, c2, false, false);
        }
        const int n  = wid * 16 + (lane & 15);
        const float bb = a.b2[n];
#pragma unroll
        for (int r = 0; r < 8; ++r) {
            const int m  = r + khb;
            const int rg = row0 + m;
            if (rg < a.R) {
                const float v = c2[r] + bb;
                if (a.scat) {
                    atomicAdd(&a.out[(long)a.scat[rg] * 128 + n], v);
                } else {
                    a.out[(long)rg * 128 + n] = v;
                }
            }
        }
    } else {
        // ---- dout == 1: scalar dot with raw fp32 W2 ----
        if (tid < 16) {
            const int rg = row0 + tid;
            if (rg < a.R) {
                float s = a.b2[0];
                for (int k2 = 0; k2 < 128; ++k2)
                    s += (float)ldsH[tid * 128 + k2] * a.w2raw[k2];
                a.out[rg] = s;
            }
        }
    }
}

// ---------------------------------------------------------------------------
// Pack fp32 weight matrix (din x dout, row-major) into WMMA B fragments.
// B-fragment layout (matches A interleave): lanes 0-15 hold
// K=ktile*32+{0..7,16..23}, lanes 16-31 hold K=ktile*32+{8..15,24..31},
// column N = lane&15 -> k = (h&7) + 8*(lane>=16) + 16*(h>=8).
// Fragment storage: [(ntile*KT + ktile)*32 + lane]*16 + h  (v16h per lane).
// ---------------------------------------------------------------------------
__global__ void pack_w(const float* W, int din, int dout, int dinp, int doutp,
                       _Float16* out) {
    const int NT = doutp >> 4, KT = dinp >> 5;
    const int total = NT * KT * 32 * 16;
    for (int i = blockIdx.x * blockDim.x + threadIdx.x; i < total;
         i += gridDim.x * blockDim.x) {
        const int h    = i & 15;
        const int lane = (i >> 4) & 31;
        const int t    = i >> 9;          // ntile*KT + ktile
        const int ktile = t % KT;
        const int ntile = t / KT;
        const int k  = (h & 7) + ((lane >> 4) << 3) + ((h >> 3) << 4);
        const int Kg = ktile * 32 + k;
        const int Ng = ntile * 16 + (lane & 15);
        const float v = (Kg < din && Ng < dout) ? W[(long)Kg * dout + Ng] : 0.0f;
        out[i] = (_Float16)v;
    }
}

// ---------------------------------------------------------------------------
// Small utility kernels
// ---------------------------------------------------------------------------
__global__ void zero_f32(float* p, long n) {
    long i = (long)blockIdx.x * blockDim.x + threadIdx.x;
    if (i < n) p[i] = 0.0f;
}

__global__ void count_idx(const int* idx, int n, float* cnt) {
    int i = blockIdx.x * blockDim.x + threadIdx.x;
    if (i < n) atomicAdd(&cnt[idx[i]], 1.0f);
}

__global__ void div_rows(float* x, const float* cnt, long n, int w) {
    long i = (long)blockIdx.x * blockDim.x + threadIdx.x;
    if (i < n) {
        float c = cnt[i / w];
        x[i] /= (c > 1.0f ? c : 1.0f);
    }
}

__global__ void scatter_add_rows(const float* src, const int* idx, long n, int w,
                                 float* dst) {
    long i = (long)blockIdx.x * blockDim.x + threadIdx.x;
    if (i < n) atomicAdd(&dst[(long)idx[i / w] * w + (i % w)], src[i]);
}

__global__ void final_logits(const float* e, float* out, int n) {
    int i = blockIdx.x * blockDim.x + threadIdx.x;
    if (i < n) out[i] = 0.5f * (e[i] + e[i + n]);
}

// ---------------------------------------------------------------------------
// Host orchestration
// ---------------------------------------------------------------------------
static inline int pad32i(int x) { return (x + 31) & ~31; }

extern "C" void kernel_launch(void* const* d_in, const int* in_sizes, int n_in,
                              void* d_out, int out_size, void* d_ws, size_t ws_size,
                              hipStream_t stream) {
    (void)n_in; (void)ws_size;
    const int NE = 400000, NN = 50000, NB = 64;

    const float* X0    = (const float*)d_in[0];
    const int*   ei    = (const int*)d_in[1];
    const float* e0    = (const float*)d_in[2];
    const float* u0    = (const float*)d_in[3];
    const int*   batch = (const int*)d_in[4];
    const int*   row   = ei;
    const int*   col   = ei + NE;

    // ---- workspace carve-out ----
    uintptr_t base = (uintptr_t)d_ws;
    auto alloc = [&](size_t bytes) -> void* {
        void* p = (void*)base;
        base += (bytes + 255) & ~(size_t)255;
        return p;
    };
    float* eA   = (float*)alloc((size_t)NE * 128 * 4);
    float* eB   = (float*)alloc((size_t)NE * 128 * 4);
    float* XA   = (float*)alloc((size_t)NN * 128 * 4);
    float* XB   = (float*)alloc((size_t)NN * 128 * 4);
    float* uA   = (float*)alloc((size_t)NB * 128 * 4);
    float* uB   = (float*)alloc((size_t)NB * 128 * 4);
    float* agg  = (float*)alloc((size_t)NN * 128 * 4);
    float* gsum = (float*)alloc((size_t)NB * 128 * 4);
    float* cntc = (float*)alloc((size_t)NN * 4);
    float* cntb = (float*)alloc((size_t)NB * 4);

    // MLP order matches jax dict-key sort: edge, glob, node1, node2
    static const int DIN[3][4]  = {{48, 136, 144, 152},
                                   {512, 256, 256, 384},
                                   {512, 129, 129, 384}};
    static const int DOUT[3][4] = {{128, 128, 128, 128},
                                   {128, 128, 128, 128},
                                   {1, 1, 128, 1}};

    _Float16* w1p[3][4];
    _Float16* w2p[3][4];
    for (int l = 0; l < 3; ++l)
        for (int m = 0; m < 4; ++m) {
            const int dinp = pad32i(DIN[l][m]);
            w1p[l][m] = (_Float16*)alloc((size_t)dinp * 128 * 2);
            w2p[l][m] = (DOUT[l][m] == 128) ? (_Float16*)alloc((size_t)128 * 128 * 2)
                                            : nullptr;
        }

    // ---- pack weights into WMMA B fragment layout ----
    for (int l = 0; l < 3; ++l)
        for (int m = 0; m < 4; ++m) {
            const int bi   = 5 + l * 16 + m * 4;   // w1,b1,w2,b2
            const int din  = DIN[l][m];
            const int dinp = pad32i(din);
            int total = (128 / 16) * (dinp / 32) * 512;
            pack_w<<<(total + 255) / 256, 256, 0, stream>>>(
                (const float*)d_in[bi], din, 128, dinp, 128, w1p[l][m]);
            if (DOUT[l][m] == 128) {
                total = (128 / 16) * (128 / 32) * 512;
                pack_w<<<(total + 255) / 256, 256, 0, stream>>>(
                    (const float*)d_in[bi + 2], 128, 128, 128, 128, w2p[l][m]);
            }
        }

    // ---- degree / per-graph counts (constant across layers) ----
    zero_f32<<<(NN + 255) / 256, 256, 0, stream>>>(cntc, NN);
    count_idx<<<(NE + 255) / 256, 256, 0, stream>>>(col, NE, cntc);
    zero_f32<<<(NB + 255) / 256, 256, 0, stream>>>(cntb, NB);
    count_idx<<<(NN + 255) / 256, 256, 0, stream>>>(batch, NN, cntb);

    // ---- layer loop ----
    const float* Xc = X0; int wX = 16;
    const float* ec = e0; int wE = 8;
    const float* uc = u0; int wU = 8;
    float* Xn = XA; float* en = eA; float* un = uA;

    for (int l = 0; l < 3; ++l) {
        // ---- EdgeModel: cat[X[row], X[col], e, u[batch[row]]] -> MLP -> en ----
        {
            MlpArgs A{};
            A.seg[0] = {Xc, row, nullptr, wX};
            A.seg[1] = {Xc, col, nullptr, wX};
            A.seg[2] = {ec, nullptr, nullptr, wE};
            A.seg[3] = {uc, row, batch, wU};
            A.nseg = 4;
            A.din  = 2 * wX + wE + wU;
            A.dinp = pad32i(A.din);
            A.dout = DOUT[l][0];
            const int bi = 5 + l * 16 + 0 * 4;
            A.w1p = w1p[l][0]; A.b1 = (const float*)d_in[bi + 1];
            A.w2p = w2p[l][0]; A.w2raw = (const float*)d_in[bi + 2];
            A.b2  = (const float*)d_in[bi + 3];
            A.out = en; A.scat = nullptr; A.R = NE;
            mlp_kernel<<<(NE + 15) / 16, NTHREADS, 0, stream>>>(A);
        }
        const int wEn = DOUT[l][0];

        // ---- NodeModel part 1: MLP(cat[X[row], en]) scatter-mean by col ----
        zero_f32<<<((long)NN * 128 + 255) / 256, 256, 0, stream>>>(agg, (long)NN * 128);
        {
            MlpArgs A{};
            A.seg[0] = {Xc, row, nullptr, wX};
            A.seg[1] = {en, nullptr, nullptr, wEn};
            A.nseg = 2;
            A.din  = wX + wEn;
            A.dinp = pad32i(A.din);
            A.dout = 128;                           // node1 dout is always H
            const int bi = 5 + l * 16 + 2 * 4;
            A.w1p = w1p[l][2]; A.b1 = (const float*)d_in[bi + 1];
            A.w2p = w2p[l][2]; A.w2raw = (const float*)d_in[bi + 2];
            A.b2  = (const float*)d_in[bi + 3];
            A.out = agg; A.scat = col; A.R = NE;    // fused atomic scatter-add
            mlp_kernel<<<(NE + 15) / 16, NTHREADS, 0, stream>>>(A);
        }
        div_rows<<<((long)NN * 128 + 255) / 256, 256, 0, stream>>>(
            agg, cntc, (long)NN * 128, 128);

        // ---- NodeModel part 2: MLP(cat[X, agg, u[batch]]) -> Xn ----
        {
            MlpArgs A{};
            A.seg[0] = {Xc, nullptr, nullptr, wX};
            A.seg[1] = {agg, nullptr, nullptr, 128};
            A.seg[2] = {uc, batch, nullptr, wU};
            A.nseg = 3;
            A.din  = wX + 128 + wU;
            A.dinp = pad32i(A.din);
            A.dout = DOUT[l][3];
            const int bi = 5 + l * 16 + 3 * 4;
            A.w1p = w1p[l][3]; A.b1 = (const float*)d_in[bi + 1];
            A.w2p = w2p[l][3]; A.w2raw = (const float*)d_in[bi + 2];
            A.b2  = (const float*)d_in[bi + 3];
            A.out = Xn; A.scat = nullptr; A.R = NN;
            mlp_kernel<<<(NN + 15) / 16, NTHREADS, 0, stream>>>(A);
        }
        const int wXn = DOUT[l][3];

        // ---- GlobalModel: MLP(cat[u, mean_g(Xn)]) -> un ----
        zero_f32<<<((long)NB * 128 + 255) / 256, 256, 0, stream>>>(gsum, (long)NB * 128);
        scatter_add_rows<<<((long)NN * wXn + 255) / 256, 256, 0, stream>>>(
            Xn, batch, (long)NN * wXn, wXn, gsum);
        div_rows<<<((long)NB * wXn + 255) / 256, 256, 0, stream>>>(
            gsum, cntb, (long)NB * wXn, wXn);
        {
            MlpArgs A{};
            A.seg[0] = {uc, nullptr, nullptr, wU};
            A.seg[1] = {gsum, nullptr, nullptr, wXn};
            A.nseg = 2;
            A.din  = wU + wXn;
            A.dinp = pad32i(A.din);
            A.dout = DOUT[l][1];
            const int bi = 5 + l * 16 + 1 * 4;
            A.w1p = w1p[l][1]; A.b1 = (const float*)d_in[bi + 1];
            A.w2p = w2p[l][1]; A.w2raw = (const float*)d_in[bi + 2];
            A.b2  = (const float*)d_in[bi + 3];
            A.out = un; A.scat = nullptr; A.R = NB;
            mlp_kernel<<<(NB + 15) / 16, NTHREADS, 0, stream>>>(A);
        }
        const int wUn = DOUT[l][1];

        // ---- ping-pong ----
        Xc = Xn; wX = wXn;
        ec = en; wE = wEn;
        uc = un; wU = wUn;
        Xn = (Xc == XA) ? XB : XA;
        en = (ec == eA) ? eB : eA;
        un = (uc == uA) ? uB : uA;
    }

    // ---- logits = (e[:n] + e[n:]) / 2, e is width-1 after layer 2 ----
    const int half = NE / 2;
    final_logits<<<(half + 255) / 256, 256, 0, stream>>>(ec, (float*)d_out, half);
    (void)out_size;
}
